// SlidingTileAttention2D_70265664962744
// MI455X (gfx1250) — matrix operand order repaired and checked by reading, not compiled, not run on hardware
//
#include <hip/hip_runtime.h>
#include <hip/hip_bf16.h>

typedef unsigned short u16;
typedef __attribute__((ext_vector_type(16))) __bf16 v16bf;
typedef __attribute__((ext_vector_type(8)))  float  v8f;

static constexpr int HID = 2048;
static constexpr int H   = 32;
static constexpr int D   = 64;
static constexpr int S   = 4096;
static constexpr int NTH = 8;
static constexpr int NTW = 4;

// ---------------------------------------------------------------- async LDS
#if __has_builtin(__builtin_amdgcn_global_load_async_to_lds_b128)
#define ASYNC_LDS 1
#else
#define ASYNC_LDS 0
#endif

#if __has_builtin(__builtin_amdgcn_s_wait_asynccnt)
#define WAIT_ASYNC() __builtin_amdgcn_s_wait_asynccnt(0)
#else
#define WAIT_ASYNC() asm volatile("s_wait_asynccnt 0x0" ::: "memory")
#endif

#if ASYNC_LDS
typedef int v4i __attribute__((vector_size(16)));
typedef __attribute__((address_space(1))) v4i glb_v4i;
typedef __attribute__((address_space(3))) v4i lds_v4i;
__device__ __forceinline__ void cp_async16(void* lds, const void* g) {
    __builtin_amdgcn_global_load_async_to_lds_b128((glb_v4i*)g, (lds_v4i*)lds,
                                                   0, 0);
}
#endif

// ---------------------------------------------------------------- helpers
__device__ __forceinline__ u16 f2bf(float f) {
    unsigned u = __float_as_uint(f);
    u += 0x7FFFu + ((u >> 16) & 1u);          // round-to-nearest-even
    return (u16)(u >> 16);
}

__device__ __forceinline__ v16bf load_frag2(const u16* p0, const u16* p1) {
    union { uint4 u[2]; v16bf v; } f;
    f.u[0] = *(const uint4*)p0;
    f.u[1] = *(const uint4*)p1;
    return f.v;
}

__device__ __forceinline__ v8f vzero8() {
    v8f z;
#pragma unroll
    for (int i = 0; i < 8; ++i) z[i] = 0.0f;
    return z;
}

__device__ __forceinline__ v8f wmma_bf16(v16bf a, v16bf b, v8f c) {
    return __builtin_amdgcn_wmma_f32_16x16x32_bf16(false, a, false, b,
                                                   (short)0, c, false, false);
}

__device__ __forceinline__ float hmax16(float v) {
#pragma unroll
    for (int m = 8; m >= 1; m >>= 1) v = fmaxf(v, __shfl_xor(v, m, 32));
    return v;
}
__device__ __forceinline__ float hsum16(float v) {
#pragma unroll
    for (int m = 8; m >= 1; m >>= 1) v += __shfl_xor(v, m, 32);
    return v;
}

// ---------------------------------------------------------------- converts
__global__ void f32_to_bf16_kernel(const float* __restrict__ src,
                                   u16* __restrict__ dst, int n) {
    int i = blockIdx.x * blockDim.x + threadIdx.x;
    if (i < n) dst[i] = f2bf(src[i]);
}

// src[R][C] f32  ->  dst[C][R] bf16   (tiled, both sides coalesced)
__global__ void transp_f32_to_bf16(const float* __restrict__ src,
                                   u16* __restrict__ dst, int R, int C) {
    __shared__ float tile[32][33];
    const int c0 = blockIdx.x * 32;
    const int r0 = blockIdx.y * 32;
    const int tx = threadIdx.x;          // 0..31
    const int ty = threadIdx.y;          // 0..7
#pragma unroll
    for (int i = 0; i < 4; ++i)
        tile[ty + i * 8][tx] = src[(size_t)(r0 + ty + i * 8) * C + c0 + tx];
    __syncthreads();
#pragma unroll
    for (int i = 0; i < 4; ++i)
        dst[(size_t)(c0 + ty + i * 8) * R + r0 + tx] =
            f2bf(tile[tx][ty + i * 8]);
}

// ---------------------------------------------------------------- GEMM (TN)
// C[M,N] = A[M,K] (bf16 row-major) * Bt[N,K]^T (weights pre-transposed)
// 256 threads (8 waves), 128x128 tile, K-step 32, double-buffered LDS,
// one barrier per step; async global->LDS DMA overlapped with WMMA.
template <bool OUT_BF16>
__global__ void __launch_bounds__(256)
gemm128_bf16_tn(const u16* __restrict__ A, const u16* __restrict__ Bt,
                void* __restrict__ Cout, int N, int K) {
    __shared__ __align__(16) u16 As[2][128 * 40];   // [row][k] stride 40
    __shared__ __align__(16) u16 Bs[2][128 * 40];   // [n][k]   stride 40

    const int tid  = threadIdx.x;
    const int wave = tid >> 5;
    const int lane = tid & 31;
    const int ln   = lane & 15;
    const int hi   = lane >> 4;
    const int wave_m = wave >> 1;                   // 0..3 (32 rows)
    const int wave_n = wave & 1;                    // 0..1 (64 cols)

    const size_t blockM = (size_t)blockIdx.y * 128;
    const size_t blockN = (size_t)blockIdx.x * 128;

    const int row = tid >> 1;                       // 0..127
    const int seg = (tid & 1) * 16;                 // 0 or 16
    const u16* aSrc = A  + (blockM + row) * (size_t)K + seg;
    const u16* bSrc = Bt + (blockN + row) * (size_t)K + seg;
    u16* aDst[2] = { &As[0][row * 40 + seg], &As[1][row * 40 + seg] };
    u16* bDst[2] = { &Bs[0][row * 40 + seg], &Bs[1][row * 40 + seg] };

    v8f acc[2][4];
#pragma unroll
    for (int mt = 0; mt < 2; ++mt)
#pragma unroll
        for (int nt = 0; nt < 4; ++nt) acc[mt][nt] = vzero8();

    const int nsteps = K >> 5;

#if ASYNC_LDS
    // issue 4 async 16B copies for one K-step
    auto issue = [&](int buf, int step) {
        const u16* ap = aSrc + (size_t)step * 32;
        const u16* bp = bSrc + (size_t)step * 32;
        cp_async16(aDst[buf], ap);  cp_async16(aDst[buf] + 8, ap + 8);
        cp_async16(bDst[buf], bp);  cp_async16(bDst[buf] + 8, bp + 8);
    };
    issue(0, 0);
#else
    uint4 ra0, ra1, rb0, rb1;
    ra0 = ((const uint4*)aSrc)[0];
    ra1 = ((const uint4*)aSrc)[1];
    rb0 = ((const uint4*)bSrc)[0];
    rb1 = ((const uint4*)bSrc)[1];
    ((uint4*)aDst[0])[0] = ra0;  ((uint4*)(aDst[0] + 8))[0] = ra1;
    ((uint4*)bDst[0])[0] = rb0;  ((uint4*)(bDst[0] + 8))[0] = rb1;
#endif

    for (int it = 0; it < nsteps; ++it) {
        const bool more = (it + 1) < nsteps;

#if ASYNC_LDS
        WAIT_ASYNC();                               // my buf[it&1] DMA landed
        __syncthreads();                            // everyone's landed; other
                                                    // buffer no longer read
        if (more) issue((it + 1) & 1, it + 1);      // DMA next behind WMMAs
#else
        if (more) {                                 // prefetch next step
            const u16* ap = aSrc + (size_t)(it + 1) * 32;
            const u16* bp = bSrc + (size_t)(it + 1) * 32;
            ra0 = ((const uint4*)ap)[0];
            ra1 = ((const uint4*)ap)[1];
            rb0 = ((const uint4*)bp)[0];
            rb1 = ((const uint4*)bp)[1];
        }
        __syncthreads();                            // buf[it&1] ready
#endif

        const u16* asb = As[it & 1];
        const u16* bsb = Bs[it & 1];
        v16bf aF[2];
#pragma unroll
        for (int mt = 0; mt < 2; ++mt) {
            const u16* p = &asb[(wave_m * 32 + mt * 16 + ln) * 40 + hi * 8];
            aF[mt] = load_frag2(p, p + 16);
        }
#pragma unroll
        for (int nt = 0; nt < 4; ++nt) {
            const u16* p = &bsb[(wave_n * 64 + nt * 16 + ln) * 40 + hi * 8];
            v16bf bF = load_frag2(p, p + 16);
#pragma unroll
            for (int mt = 0; mt < 2; ++mt)
                acc[mt][nt] = wmma_bf16(aF[mt], bF, acc[mt][nt]);
        }

#if !ASYNC_LDS
        if (more) {                                 // stage next into other buf
            u16* ad = aDst[(it + 1) & 1];
            u16* bd = bDst[(it + 1) & 1];
            ((uint4*)ad)[0] = ra0;  ((uint4*)(ad + 8))[0] = ra1;
            ((uint4*)bd)[0] = rb0;  ((uint4*)(bd + 8))[0] = rb1;
        }
#endif
    }

#pragma unroll
    for (int mt = 0; mt < 2; ++mt)
#pragma unroll
        for (int nt = 0; nt < 4; ++nt)
#pragma unroll
            for (int r = 0; r < 8; ++r) {
                size_t orow = blockM + wave_m * 32 + mt * 16 + r + 8 * hi;
                size_t ocol = blockN + wave_n * 64 + nt * 16 + ln;
                float v = acc[mt][nt][r];
                if (OUT_BF16) ((u16*)Cout)[orow * N + ocol] = f2bf(v);
                else          ((float*)Cout)[orow * N + ocol] = v;
            }
}

// ---------------------------------------------------------------- attention
// One workgroup per (b, head, query tile). 8 waves; wave w owns the 16 query
// rows with th == w. Flash-style online softmax over 4 clamped-window KV tiles.
__global__ void __launch_bounds__(256)
sta2d_attn(const u16* __restrict__ Q, const u16* __restrict__ K,
           const u16* __restrict__ V, u16* __restrict__ O) {
    __shared__ __align__(16) u16 bufA[128 * 136];  // K tile (cols 0..63), then P
    __shared__ __align__(16) u16 bufV[64 * 136];   // V transposed [dim][token]

    const int tid  = threadIdx.x;
    const int wave = tid >> 5;
    const int lane = tid & 31;
    const int ln   = lane & 15;
    const int hi   = lane >> 4;

    const int bid = blockIdx.x;
    const int t   = bid & 31;
    const int h   = (bid >> 5) & 31;
    const int b   = bid >> 10;
    const int nth = t >> 2;
    const int ntw = t & 3;
    const int ch  = min(max(nth, 1), NTH - 1);
    const int cw  = min(max(ntw, 1), NTW - 1);

    // Q strip (16 rows) as two A-fragments, straight from global.
    const size_t qrow = (size_t)b * S + nth * 512 + wave * 64 + ntw * 16 + ln;
    const u16* qp = Q + qrow * HID + h * D;
    v16bf a_q[2];
    a_q[0] = load_frag2(qp +      hi * 8, qp + 16 + hi * 8);
    a_q[1] = load_frag2(qp + 32 + hi * 8, qp + 48 + hi * 8);

    float m_i[8], l_i[8];
    v8f   o_acc[4];
#pragma unroll
    for (int r = 0; r < 8; ++r) { m_i[r] = -1e30f; l_i[r] = 0.0f; }
#pragma unroll
    for (int c = 0; c < 4; ++c) o_acc[c] = vzero8();

    for (int kv = 0; kv < 4; ++kv) {
        const int kr = ch - 1 + (kv >> 1);
        const int kc = cw - 1 + (kv & 1);

        __syncthreads();   // prior reads of bufA / bufV finished
        {   // cooperative load: 2 threads per token, 32 dims each
            const int tok  = tid >> 1;
            const int half = tid & 1;
            const size_t srow =
                (size_t)b * S + kr * 512 + (tok >> 4) * 64 + kc * 16 + (tok & 15);
            const u16* kp = K + srow * HID + h * D + half * 32;
            u16* dk = &bufA[tok * 136 + half * 32];
#if ASYNC_LDS
            cp_async16(dk,      kp);
            cp_async16(dk + 8,  kp + 8);
            cp_async16(dk + 16, kp + 16);
            cp_async16(dk + 24, kp + 24);
#else
            ((uint4*)dk)[0] = ((const uint4*)kp)[0];
            ((uint4*)dk)[1] = ((const uint4*)kp)[1];
            ((uint4*)dk)[2] = ((const uint4*)kp)[2];
            ((uint4*)dk)[3] = ((const uint4*)kp)[3];
#endif
            const u16* vp = V + srow * HID + h * D + half * 32;
            union { uint4 u[4]; u16 s[32]; } vt;
            vt.u[0] = ((const uint4*)vp)[0];
            vt.u[1] = ((const uint4*)vp)[1];
            vt.u[2] = ((const uint4*)vp)[2];
            vt.u[3] = ((const uint4*)vp)[3];
#pragma unroll
            for (int d = 0; d < 32; ++d)
                bufV[(half * 32 + d) * 136 + tok] = vt.s[d];
        }
#if ASYNC_LDS
        WAIT_ASYNC();
#endif
        __syncthreads();

        // scores: 16x128 strip = 8 tiles, K = 64 -> 2 WMMAs each
        v8f sc[8];
#pragma unroll
        for (int nt = 0; nt < 8; ++nt) {
            const u16* krow = &bufA[(nt * 16 + ln) * 136];
            v16bf bk0 = load_frag2(krow + hi * 8,      krow + hi * 8 + 16);
            v16bf bk1 = load_frag2(krow + 32 + hi * 8, krow + 32 + hi * 8 + 16);
            v8f s = wmma_bf16(a_q[0], bk0, vzero8());
            sc[nt] = wmma_bf16(a_q[1], bk1, s);
        }

        // online softmax (rows live per element-r, replicated over 16-lane half)
        float fac[8];
#pragma unroll
        for (int r = 0; r < 8; ++r) {
            float mx = -1e30f;
#pragma unroll
            for (int nt = 0; nt < 8; ++nt) {
                sc[nt][r] *= 0.125f;                 // 1/sqrt(D)
                mx = fmaxf(mx, sc[nt][r]);
            }
            mx = hmax16(mx);
            float mnew = fmaxf(m_i[r], mx);
            fac[r] = __expf(m_i[r] - mnew);
            float rs = 0.0f;
#pragma unroll
            for (int nt = 0; nt < 8; ++nt) {
                float p = __expf(sc[nt][r] - mnew);
                sc[nt][r] = p;
                rs += p;
            }
            rs = hsum16(rs);
            l_i[r] = l_i[r] * fac[r] + rs;
            m_i[r] = mnew;
        }

        __syncthreads();   // everyone done reading K -> reuse bufA for P

        // write P (bf16) to this wave's private 16x128 region of bufA
#pragma unroll
        for (int nt = 0; nt < 8; ++nt)
#pragma unroll
            for (int r = 0; r < 8; ++r)
                bufA[(wave * 16 + r + 8 * hi) * 136 + nt * 16 + ln] =
                    f2bf(sc[nt][r]);

        // rescale running output
#pragma unroll
        for (int c = 0; c < 4; ++c)
#pragma unroll
            for (int r = 0; r < 8; ++r) o_acc[c][r] *= fac[r];

        // P (16x128) x V (128x64): re-load P as A-fragments, V^T B-fragments
        v16bf aP[4];
#pragma unroll
        for (int kk = 0; kk < 4; ++kk) {
            const u16* p = &bufA[(wave * 16 + ln) * 136 + kk * 32 + hi * 8];
            aP[kk] = load_frag2(p, p + 16);
        }
#pragma unroll
        for (int c = 0; c < 4; ++c) {
            const u16* vrow = &bufV[(c * 16 + ln) * 136];
#pragma unroll
            for (int kk = 0; kk < 4; ++kk) {
                v16bf bV = load_frag2(vrow + kk * 32 + hi * 8,
                                      vrow + kk * 32 + hi * 8 + 16);
                o_acc[c] = wmma_bf16(aP[kk], bV, o_acc[c]);
            }
        }
    }

    // normalize + store (seq-major layout, bf16)
#pragma unroll
    for (int c = 0; c < 4; ++c)
#pragma unroll
        for (int r = 0; r < 8; ++r) {
            int tw = r + 8 * hi;
            size_t grow = (size_t)b * S + nth * 512 + wave * 64 + ntw * 16 + tw;
            O[grow * HID + h * D + c * 16 + ln] = f2bf(o_acc[c][r] / l_i[r]);
        }
}

// ---------------------------------------------------------------- launch
extern "C" void kernel_launch(void* const* d_in, const int* in_sizes, int n_in,
                              void* d_out, int out_size, void* d_ws,
                              size_t ws_size, hipStream_t stream) {
    const float* X  = (const float*)d_in[0];
    const float* Wq = (const float*)d_in[1];
    const float* Wk = (const float*)d_in[2];
    const float* Wv = (const float*)d_in[3];
    const float* Wo = (const float*)d_in[4];

    const size_t NX = (size_t)2 * S * HID;   // 16,777,216 elems
    const size_t NW = (size_t)HID * HID;     //  4,194,304 elems

    u16* ws  = (u16*)d_ws;
    u16* Xb  = ws;                           // also reused as O (bf16)
    u16* Wqt = ws + NX;                      // weights stored transposed [N][K]
    u16* Wkt = ws + NX + NW;
    u16* Wvt = ws + NX + 2 * NW;
    u16* Wot = ws + NX + 3 * NW;
    u16* Qb  = ws + NX + 4 * NW;
    u16* Kb  = Qb + NX;
    u16* Vb  = Kb + NX;

    f32_to_bf16_kernel<<<(int)((NX + 255) / 256), 256, 0, stream>>>(X, Xb, (int)NX);

    dim3 tgrid(HID / 32, HID / 32), tblk(32, 8);
    transp_f32_to_bf16<<<tgrid, tblk, 0, stream>>>(Wq, Wqt, HID, HID);
    transp_f32_to_bf16<<<tgrid, tblk, 0, stream>>>(Wk, Wkt, HID, HID);
    transp_f32_to_bf16<<<tgrid, tblk, 0, stream>>>(Wv, Wvt, HID, HID);
    transp_f32_to_bf16<<<tgrid, tblk, 0, stream>>>(Wo, Wot, HID, HID);

    dim3 ggrid(HID / 128, (2 * S) / 128);    // (16, 64)
    gemm128_bf16_tn<true><<<ggrid, 256, 0, stream>>>(Xb, Wqt, Qb, HID, HID);
    gemm128_bf16_tn<true><<<ggrid, 256, 0, stream>>>(Xb, Wkt, Kb, HID, HID);
    gemm128_bf16_tn<true><<<ggrid, 256, 0, stream>>>(Xb, Wvt, Vb, HID, HID);

    sta2d_attn<<<2 * H * (NTH * NTW), 256, 0, stream>>>(Qb, Kb, Vb, Xb);

    gemm128_bf16_tn<false><<<ggrid, 256, 0, stream>>>(Xb, Wot, d_out, HID, HID);
}
